// TextGenerationModel_12635793785045
// MI455X (gfx1250) — matrix operand order repaired and checked by reading, not compiled, not run on hardware
//
#include <hip/hip_runtime.h>

// ---- model dims ----
#define BB   64
#define SS   512
#define VV   128
#define HH   1024
#define GG   (4*HH)          // 4096
#define BH   (BB*HH)         // 65536

typedef __attribute__((ext_vector_type(16))) __bf16 v16bf;
typedef __attribute__((ext_vector_type(8)))  __bf16 v8bf;
typedef __attribute__((ext_vector_type(8)))  float  v8f;
typedef __attribute__((ext_vector_type(4)))  float  v4f;

__device__ inline v8f vzero8() {
    v8f z;
#pragma unroll
    for (int i = 0; i < 8; ++i) z[i] = 0.0f;
    return z;
}

// A-tile (16x32 bf16) per-lane load, contiguous-k order shared with load_b_bf:
// lane l: row m = l&15 ; contiguous K chunk [2*ka..2*ka+15], ka = (l<16)?0:8
__device__ inline v16bf load_a_bf(const __bf16* __restrict__ row, int k0, int ka) {
    v8bf lo = *(const v8bf*)(row + k0 + 2 * ka);
    v8bf hi = *(const v8bf*)(row + k0 + 2 * ka + 8);
    return __builtin_shufflevector(lo, hi, 0,1,2,3,4,5,6,7,8,9,10,11,12,13,14,15);
}

// same A layout but source is fp32 (converted in-register to bf16)
__device__ inline v8bf cvt8(const float* __restrict__ p) {
    const v4f a = *(const v4f*)(p);
    const v4f b = *(const v4f*)(p + 4);
    v8bf r;
#pragma unroll
    for (int i = 0; i < 4; ++i) { r[i] = (__bf16)a[i]; r[4+i] = (__bf16)b[i]; }
    return r;
}
__device__ inline v16bf load_a_f32(const float* __restrict__ row, int k0, int ka) {
    v8bf lo = cvt8(row + k0 + 2 * ka);
    v8bf hi = cvt8(row + k0 + 2 * ka + 8);
    return __builtin_shufflevector(lo, hi, 0,1,2,3,4,5,6,7,8,9,10,11,12,13,14,15);
}

// B-tile (32x16 bf16): lane l: col n = l&15, contiguous K chunk of 16 at kb=(l<16)?0:16.
// W stored (N rows x K cols) row-major -> one 32-byte load from row n.
__device__ inline v16bf load_b_bf(const __bf16* __restrict__ wrow, int k0, int kb) {
    return *(const v16bf*)(wrow + k0 + kb);
}

__device__ inline float sigm(float x) { return 1.0f / (1.0f + __expf(-x)); }

// ---------------- prep: cast weights to bf16, fuse biases ----------------
__global__ void prep_kernel(const float* __restrict__ Wih0, const float* __restrict__ Whh0,
                            const float* __restrict__ Wih1, const float* __restrict__ Whh1,
                            const float* __restrict__ Wfc,
                            const float* __restrict__ bih0, const float* __restrict__ bhh0,
                            const float* __restrict__ bih1, const float* __restrict__ bhh1,
                            __bf16* wih0b, __bf16* whh0b, __bf16* wih1b, __bf16* whh1b,
                            __bf16* wfcb, float* bias0, float* bias1) {
    const long n0 = (long)GG*VV;
    const long n1 = n0 + (long)GG*HH;
    const long n2 = n1 + (long)GG*HH;
    const long n3 = n2 + (long)GG*HH;
    const long n4 = n3 + (long)VV*HH;
    const long n5 = n4 + GG;
    const long n6 = n5 + GG;
    const long stride = (long)gridDim.x * blockDim.x;
    for (long i = (long)blockIdx.x * blockDim.x + threadIdx.x; i < n6; i += stride) {
        if      (i < n0) wih0b[i]      = (__bf16)Wih0[i];
        else if (i < n1) whh0b[i - n0] = (__bf16)Whh0[i - n0];
        else if (i < n2) wih1b[i - n1] = (__bf16)Wih1[i - n1];
        else if (i < n3) whh1b[i - n2] = (__bf16)Whh1[i - n2];
        else if (i < n4) wfcb[i - n3]  = (__bf16)Wfc[i - n3];
        else if (i < n5) { long j = i - n4; bias0[j] = bih0[j] + bhh0[j]; }
        else             { long j = i - n5; bias1[j] = bih1[j] + bhh1[j]; }
    }
}

// ---------------- software-pipelined dual-layer LSTM step ----------------
// One launch computes layer0 @ t0 AND layer1 @ t1 = t0-1 concurrently:
//   blocks [0,128)   -> layer 0, blocks [128,256) -> layer 1.
// Per layer: 256 output tiles (4 batch-tiles x 64 hidden-tiles), each tile
// owned by 4 waves (one per gate i/f/g/o). Each wave accumulates its gate's
// full-K dot products with WMMA, gates are recombined through LDS, and the
// gate-0 wave performs the fused LSTM cell update. 2048 waves per launch.
__global__ __launch_bounds__(256) void lstm_pipe_kernel(
    const float*  __restrict__ x,      // (B, S*V) fp32
    __bf16*       __restrict__ h1,     // (S,B,H) bf16 : layer0 out / layer1 in
    __bf16*       __restrict__ h2,     // (S,B,H) bf16 : layer1 out
    const __bf16* __restrict__ wih0,   // (4H, V)
    const __bf16* __restrict__ whh0,   // (4H, H)
    const __bf16* __restrict__ wih1,   // (4H, H)
    const __bf16* __restrict__ whh1,   // (4H, H)
    const float*  __restrict__ bias0,  // (4H)
    const float*  __restrict__ bias1,  // (4H)
    float*        __restrict__ c0,     // (B,H)
    float*        __restrict__ c1,     // (B,H)
    int t0, int t1) {
    __shared__ float red[2][3][8][32];     // [tile][gate-1][r][lane], 6 KB

    const bool is_l0 = blockIdx.x < 128;
    const int  t     = is_l0 ? t0 : t1;
    if (t < 0 || t >= SS) return;          // uniform per block: no barrier hazard

    const int lane = threadIdx.x & 31;
    const int half = lane >> 4;
    const int ka   = half * 8;
    const int kb   = half * 16;
    const int w    = threadIdx.x >> 5;     // wave in block: 0..7
    const int tile = w >> 2;               // 0..1
    const int g    = w & 3;                // gate: 0=i 1=f 2=g 3=o
    const int tid  = (is_l0 ? blockIdx.x : blockIdx.x - 128) * 2 + tile; // 0..255
    const int mt   = tid >> 6;             // batch tile 0..3
    const int jt   = tid & 63;             // hidden tile 0..63
    const int mrow = mt * 16 + (lane & 15);
    const int ncol = jt * 16 + (lane & 15);

    const __bf16* Win   = is_l0 ? wih0 : wih1;
    const __bf16* Whh   = is_l0 ? whh0 : whh1;
    const float*  bias  = is_l0 ? bias0 : bias1;
    float*        c     = is_l0 ? c0 : c1;
    __bf16*       h_all = is_l0 ? h1 : h2;

    v8f acc = vzero8();

    // ---- input projection for this gate ----
    if (is_l0) {
        const __bf16* wrow = Win + (size_t)(g * HH + ncol) * VV;
        const float*  xrow = x + (size_t)mrow * (SS * VV) + (size_t)t * VV;
        for (int k0 = 0; k0 < VV; k0 += 32) {
            v16bf a = load_a_f32(xrow, k0, ka);
            v16bf b = load_b_bf(wrow, k0, kb);
            acc = __builtin_amdgcn_wmma_f32_16x16x32_bf16(
                false, a, false, b, (short)0, acc, false, false);
        }
    } else {
        const __bf16* wrow = Win + (size_t)(g * HH + ncol) * HH;
        const __bf16* arow = h1 + (size_t)t * BH + (size_t)mrow * HH;
        for (int k0 = 0; k0 < HH; k0 += 32) {
            v16bf a = load_a_bf(arow, k0, ka);
            v16bf b = load_b_bf(wrow, k0, kb);
            acc = __builtin_amdgcn_wmma_f32_16x16x32_bf16(
                false, a, false, b, (short)0, acc, false, false);
        }
    }

    // ---- recurrence for this gate ----
    if (t > 0) {
        const __bf16* wrow = Whh + (size_t)(g * HH + ncol) * HH;
        const __bf16* hrow = h_all + (size_t)(t - 1) * BH + (size_t)mrow * HH;
        for (int k0 = 0; k0 < HH; k0 += 32) {
            v16bf a = load_a_bf(hrow, k0, ka);
            v16bf b = load_b_bf(wrow, k0, kb);
            acc = __builtin_amdgcn_wmma_f32_16x16x32_bf16(
                false, a, false, b, (short)0, acc, false, false);
        }
    }

    // ---- recombine gates through LDS ----
    if (g != 0) {
#pragma unroll
        for (int r = 0; r < 8; ++r) red[tile][g - 1][r][lane] = acc[r];
    }
    __syncthreads();

    // ---- fused LSTM cell (gate-0 wave of each tile) ----
    if (g == 0) {
        const float bi = bias[0 * HH + ncol];
        const float bf = bias[1 * HH + ncol];
        const float bg = bias[2 * HH + ncol];
        const float bo = bias[3 * HH + ncol];
#pragma unroll
        for (int r = 0; r < 8; ++r) {
            const int brow = mt * 16 + r + 8 * half;   // C layout: VGPR r -> M = r(+8)
            const size_t ci = (size_t)brow * HH + ncol;
            const float iv = sigm(acc[r] + bi);
            const float fv = sigm(red[tile][0][r][lane] + bf);
            const float gv = tanhf(red[tile][1][r][lane] + bg);
            const float ov = sigm(red[tile][2][r][lane] + bo);
            const float cold = (t == 0) ? 0.0f : c[ci];
            const float cn = fv * cold + iv * gv;
            c[ci] = cn;
            h_all[(size_t)t * BH + ci] = (__bf16)(ov * tanhf(cn));
        }
    }
}

// ---------------- FC + log-softmax ----------------
__global__ __launch_bounds__(256) void fc_softmax_kernel(
    const __bf16* __restrict__ h2,   // (S*B, H) bf16
    const __bf16* __restrict__ Wfc,  // (V, H) bf16
    const float*  __restrict__ bfc,  // (V) fp32
    float*        __restrict__ out)  // (B,S,V) fp32
{
    const int lane = threadIdx.x & 31;
    const int half = lane >> 4;
    const int rt   = blockIdx.x * 8 + (threadIdx.x >> 5);  // row-tile 0..2047
    const int mrow = rt * 16 + (lane & 15);

    v8f acc[8];
#pragma unroll
    for (int nt = 0; nt < 8; ++nt) acc[nt] = vzero8();

    const __bf16* arow = h2 + (size_t)mrow * HH;
    for (int k0 = 0; k0 < HH; k0 += 32) {
        v16bf a = load_a_bf(arow, k0, half * 8);
#pragma unroll
        for (int nt = 0; nt < 8; ++nt) {
            const int n = nt * 16 + (lane & 15);
            v16bf b = load_b_bf(Wfc + (size_t)n * HH, k0, half * 16);
            acc[nt] = __builtin_amdgcn_wmma_f32_16x16x32_bf16(
                false, a, false, b, (short)0, acc[nt], false, false);
        }
    }

    float biasv[8];
#pragma unroll
    for (int nt = 0; nt < 8; ++nt) biasv[nt] = bfc[nt * 16 + (lane & 15)];

#pragma unroll
    for (int r = 0; r < 8; ++r) {
        float vals[8];
        float mx = -3.4e38f;
#pragma unroll
        for (int nt = 0; nt < 8; ++nt) { vals[nt] = acc[nt][r] + biasv[nt]; mx = fmaxf(mx, vals[nt]); }
        for (int off = 1; off < 16; off <<= 1) mx = fmaxf(mx, __shfl_xor(mx, off, 16));
        float s = 0.0f;
#pragma unroll
        for (int nt = 0; nt < 8; ++nt) s += __expf(vals[nt] - mx);
        for (int off = 1; off < 16; off <<= 1) s += __shfl_xor(s, off, 16);
        const float lse = mx + __logf(s);
        const int row = rt * 16 + r + 8 * half;       // row = s*B + b
        const int sq  = row >> 6;
        const int bq  = row & 63;
        float* orow = out + ((size_t)bq * SS + sq) * VV;
#pragma unroll
        for (int nt = 0; nt < 8; ++nt) orow[nt * 16 + (lane & 15)] = vals[nt] - lse;
    }
}

extern "C" void kernel_launch(void* const* d_in, const int* in_sizes, int n_in,
                              void* d_out, int out_size, void* d_ws, size_t ws_size,
                              hipStream_t stream) {
    const float* x    = (const float*)d_in[0];
    const float* Wih0 = (const float*)d_in[1];
    const float* Whh0 = (const float*)d_in[2];
    const float* bih0 = (const float*)d_in[3];
    const float* bhh0 = (const float*)d_in[4];
    const float* Wih1 = (const float*)d_in[5];
    const float* Whh1 = (const float*)d_in[6];
    const float* bih1 = (const float*)d_in[7];
    const float* bhh1 = (const float*)d_in[8];
    const float* Wfc  = (const float*)d_in[9];
    const float* bfc  = (const float*)d_in[10];
    float* out = (float*)d_out;

    char* base = (char*)d_ws;
    size_t off = 0;
    auto take = [&](size_t bytes) { char* p = base + off; off += (bytes + 255) & ~(size_t)255; return p; };
    __bf16* wih0b = (__bf16*)take((size_t)GG * VV * 2);
    __bf16* whh0b = (__bf16*)take((size_t)GG * HH * 2);
    __bf16* wih1b = (__bf16*)take((size_t)GG * HH * 2);
    __bf16* whh1b = (__bf16*)take((size_t)GG * HH * 2);
    __bf16* wfcb  = (__bf16*)take((size_t)VV * HH * 2);
    float*  bias0 = (float*)take((size_t)GG * 4);
    float*  bias1 = (float*)take((size_t)GG * 4);
    float*  c0    = (float*)take((size_t)BH * 4);
    float*  c1    = (float*)take((size_t)BH * 4);
    __bf16* h1    = (__bf16*)take((size_t)SS * BH * 2);
    __bf16* h2    = (__bf16*)take((size_t)SS * BH * 2);

    prep_kernel<<<2048, 256, 0, stream>>>(Wih0, Whh0, Wih1, Whh1, Wfc,
                                          bih0, bhh0, bih1, bhh1,
                                          wih0b, whh0b, wih1b, whh1b, wfcb, bias0, bias1);

    // Pipelined recurrence: launch tt computes layer0@tt and layer1@(tt-1).
    for (int tt = 0; tt <= SS; ++tt)
        lstm_pipe_kernel<<<256, 256, 0, stream>>>(x, h1, h2,
                                                  wih0b, whh0b, wih1b, whh1b,
                                                  bias0, bias1, c0, c1,
                                                  tt, tt - 1);

    fc_softmax_kernel<<<256, 256, 0, stream>>>(h2, wfcb, bfc, out);
}
